// VectorQuantizer_49048526520937
// MI455X (gfx1250) — matrix operand order repaired and checked by reading, not compiled, not run on hardware
//
#include <hip/hip_runtime.h>
#include <hip/hip_bf16.h>

typedef __attribute__((ext_vector_type(16))) __bf16 v16bf;
typedef __attribute__((ext_vector_type(8)))  float  v8f;

typedef unsigned short u16;
typedef unsigned int   u32;

union Frag16 {           // 16 bf16 = 32 bytes, loaded as two 16B chunks
    v16bf v;
    uint4 q[2];
};

struct RedT { float v; int k; };

__device__ __forceinline__ u16 f2bf(float f) {
    u32 u = __float_as_uint(f);
    u32 r = u + 0x7fffu + ((u >> 16) & 1u);   // round-to-nearest-even
    return (u16)(r >> 16);
}

// ---------------------------------------------------------------------------
// Kernel 1: codebook fp32 -> bf16, half-norms 0.5*||c_k||^2, zero loss slot.
// grid = (1024), block = (256): block k, thread d.
// ---------------------------------------------------------------------------
__global__ __launch_bounds__(256) void vq_prep(const float* __restrict__ cb,
                                               u16* __restrict__ cbB,
                                               float* __restrict__ hn,
                                               float* __restrict__ lossSlot) {
    const int k = blockIdx.x;
    const int d = threadIdx.x;
    float v = cb[k * 256 + d];
    cbB[k * 256 + d] = f2bf(v);
    float s = v * v;
    #pragma unroll
    for (int m = 16; m >= 1; m >>= 1) s += __shfl_xor(s, m, 32);
    __shared__ float part[8];
    if ((d & 31) == 0) part[d >> 5] = s;
    __syncthreads();
    if (d == 0) {
        float t = 0.f;
        #pragma unroll
        for (int j = 0; j < 8; ++j) t += part[j];
        hn[k] = 0.5f * t;
        if (k == 0) *lossSlot = 0.f;
    }
}

// ---------------------------------------------------------------------------
// Kernel 2: fused score GEMM (bf16 WMMA) + argmax over K.
// grid = (N/32) = 1024, block = 256 (8 waves = 2 nTiles x 4 kTiles).
// scores s[n,k] = z_n . c_k - 0.5*||c_k||^2 ; code[n] = argmax_k s[n,k].
// ---------------------------------------------------------------------------
__global__ __launch_bounds__(256) void vq_gemm_argmax(const float* __restrict__ z,
                                                      const u16* __restrict__ cbB,
                                                      const float* __restrict__ hn,
                                                      int* __restrict__ codes) {
    __shared__ __align__(16) u16 ldsA[32][264];   // 32 n-rows x 256 d (bf16), padded
    __shared__ RedT red[2][4][16];

    const int tid = threadIdx.x;
    const int n0  = blockIdx.x * 32;
    const int b   = n0 >> 10;                 // batch (n0 multiple of 32, 1024 per batch)
    const int hw0 = n0 & 1023;
    const float* zb = z + (size_t)b * 262144 + hw0;   // z[b, d, hw] = zb[d*1024 + nn]

    // Stage A tile: coalesced global reads (n-contiguous per d), convert to bf16.
    for (int i = tid; i < 32 * 256; i += 256) {
        int nn = i & 31;
        int d  = i >> 5;
        ldsA[nn][d] = f2bf(zb[d * 1024 + nn]);
    }
    __syncthreads();

    const int wave  = tid >> 5;
    const int lane  = tid & 31;
    const int nTile = wave >> 2;      // 0..1 -> n-offset 16*nTile
    const int kt    = wave & 3;       // 0..3 -> k-offset 16*kt within 64-k block
    const int lhalf = lane >> 4;      // 0 / 1
    const int l16   = lane & 15;

    // Preload all 8 A fragments (full D=256) per ISA 16-bit A layout:
    // lane L: M = L%16 ; K = {0..7,16..23} (lanes 0-15) / {8..15,24..31} (lanes 16-31)
    Frag16 afr[8];
    const int nrow = nTile * 16 + l16;
    #pragma unroll
    for (int ds = 0; ds < 8; ++ds) {
        int dA = ds * 32 + lhalf * 8;
        afr[ds].q[0] = *(const uint4*)&ldsA[nrow][dA];
        afr[ds].q[1] = *(const uint4*)&ldsA[nrow][dA + 16];
    }

    float bestV[8];
    int   bestK[8];
    #pragma unroll
    for (int v = 0; v < 8; ++v) { bestV[v] = -3.4e38f; bestK[v] = 0; }

    for (int k0 = 0; k0 < 1024; k0 += 64) {
        const int myk = k0 + kt * 16 + l16;
        // B 16-bit layout: lane L holds column N=L%16, same K order as A:
        // lanes 0-15: K={0..7,16..23} ; lanes 16-31: K={8..15,24..31} of row myk
        const u16* cbrow = cbB + (size_t)myk * 256 + lhalf * 8;

        v8f acc = {};
        #pragma unroll
        for (int ds = 0; ds < 8; ++ds) {
            Frag16 bfr;
            bfr.q[0] = *(const uint4*)(cbrow + ds * 32);
            bfr.q[1] = *(const uint4*)(cbrow + ds * 32 + 16);
            acc = __builtin_amdgcn_wmma_f32_16x16x32_bf16(
                false, afr[ds].v, false, bfr.v, (short)0, acc, false, false);
        }

        const float h = hn[myk];
        #pragma unroll
        for (int v = 0; v < 8; ++v) {
            float s = acc[v] - h;              // D slot (v,L): M=v+8*lhalf, N=l16
            bool better = (s > bestV[v]) || (s == bestV[v] && myk < bestK[v]);
            bestV[v] = better ? s   : bestV[v];
            bestK[v] = better ? myk : bestK[v];
        }
    }

    // Reduce argmax across the 16 lanes of each half (16 k-candidates each).
    #pragma unroll
    for (int v = 0; v < 8; ++v) {
        float mv = bestV[v];
        int   mk = bestK[v];
        #pragma unroll
        for (int m = 1; m < 16; m <<= 1) {
            float ov = __shfl_xor(mv, m, 32);
            int   ok = __shfl_xor(mk, m, 32);
            bool take = (ov > mv) || (ov == mv && ok < mk);
            mv = take ? ov : mv;
            mk = take ? ok : mk;
        }
        if (l16 == 0) {                       // lane 0 -> M=v, lane 16 -> M=v+8
            int M = v + 8 * lhalf;
            red[nTile][kt][M].v = mv;
            red[nTile][kt][M].k = mk;
        }
    }
    __syncthreads();

    // Final reduce across the 4 k-tile waves; write one code per n-row.
    if (tid < 32) {
        int nt = tid >> 4, M = tid & 15;
        float mv = red[nt][0][M].v;
        int   mk = red[nt][0][M].k;
        #pragma unroll
        for (int j = 1; j < 4; ++j) {
            float ov = red[nt][j][M].v;
            int   ok = red[nt][j][M].k;
            bool take = (ov > mv) || (ov == mv && ok < mk);
            mv = take ? ov : mv;
            mk = take ? ok : mk;
        }
        codes[n0 + nt * 16 + M] = mk;
    }
}

// ---------------------------------------------------------------------------
// Kernel 3: scatter z_q (fp32 codebook rows), write indices, fused vq_loss.
// grid = N/64 = 512, block = 256. vq_loss = 1.25 * mean((z_q - z_e)^2).
// ---------------------------------------------------------------------------
__global__ __launch_bounds__(256) void vq_scatter_loss(const float* __restrict__ z,
                                                       const float* __restrict__ cb,
                                                       const int* __restrict__ codes,
                                                       float* __restrict__ zq,
                                                       float* __restrict__ outIdx,
                                                       float* __restrict__ outLoss) {
    const int tid = threadIdx.x;
    const int n0  = blockIdx.x * 64;
    const int b   = n0 >> 10;
    const int hw0 = n0 & 1023;

    __shared__ int scode[64];
    if (tid < 64) {
        int c = codes[n0 + tid];
        scode[tid] = c;
        outIdx[n0 + tid] = (float)c;          // indices in flat (B,H,W) == n order
    }
    __syncthreads();

    const float* zb = z  + (size_t)b * 262144 + hw0;
    float*       qb = zq + (size_t)b * 262144 + hw0;

    float sum = 0.f;
    for (int i = tid; i < 64 * 256; i += 256) {
        int nn = i & 63;
        int d  = i >> 6;
        float q = cb[(size_t)scode[nn] * 256 + d];  // L2-resident gather
        float e = zb[d * 1024 + nn];                // coalesced
        qb[d * 1024 + nn] = q;                      // coalesced
        float diff = q - e;
        sum += diff * diff;
    }
    #pragma unroll
    for (int m = 16; m >= 1; m >>= 1) sum += __shfl_xor(sum, m, 32);
    __shared__ float part[8];
    if ((tid & 31) == 0) part[tid >> 5] = sum;
    __syncthreads();
    if (tid == 0) {
        float t = 0.f;
        #pragma unroll
        for (int j = 0; j < 8; ++j) t += part[j];
        atomicAdd(outLoss, t * (1.25f / 8388608.0f));
    }
}

// ---------------------------------------------------------------------------
extern "C" void kernel_launch(void* const* d_in, const int* in_sizes, int n_in,
                              void* d_out, int out_size, void* d_ws, size_t ws_size,
                              hipStream_t stream) {
    const float* z  = (const float*)d_in[0];   // (32, 256, 32, 32)
    const float* cb = (const float*)d_in[1];   // (1024, 256)

    float* out     = (float*)d_out;
    float* zq      = out;                      // 8388608 floats
    float* outIdx  = out + 8388608;            // 32768 floats
    float* outLoss = out + 8388608 + 32768;    // 1 float

    char* ws   = (char*)d_ws;
    u16* cbB   = (u16*)ws;                               // 512 KB bf16 codebook
    float* hn  = (float*)(ws + 1024 * 256 * 2);          // 4 KB half-norms
    int* codes = (int*)(ws + 1024 * 256 * 2 + 1024 * 4); // 128 KB codes

    vq_prep<<<1024, 256, 0, stream>>>(cb, cbB, hn, outLoss);
    vq_gemm_argmax<<<1024, 256, 0, stream>>>(z, cbB, hn, codes);
    vq_scatter_loss<<<512, 256, 0, stream>>>(z, cb, codes, zq, outIdx, outLoss);
}
